// GAT_4698694222576
// MI455X (gfx1250) — matrix operand order repaired and checked by reading, not compiled, not run on hardware
//
#include <hip/hip_runtime.h>
#include <math.h>

#define N_NODES 50000
#define N_EDGES 800000
#define NFEAT   256
#define NHID    16
#define HEADS   8
#define NCLASS  16
#define H1DIM   (HEADS * NHID)        /* 128 */
#define ETOT    (N_EDGES + N_NODES)   /* 850000 edges incl. self-loops */
#define NEG_SLOPE 0.2f

typedef __attribute__((ext_vector_type(16))) __bf16       v16bf;
typedef __attribute__((ext_vector_type(8)))  float        v8f;
typedef __attribute__((ext_vector_type(8)))  unsigned int v8u;

// ---------- helpers ----------
__device__ __forceinline__ unsigned short f2bf(float f) {
  union { float f; unsigned int u; } c; c.f = f;
  unsigned int u = c.u + 0x7FFFu + ((c.u >> 16) & 1u);   // round-to-nearest-even
  return (unsigned short)(u >> 16);
}
// monotone float<->int mapping so integer atomic max == float max
__device__ __forceinline__ int   ford(float f) { int i = __float_as_int(f); return i >= 0 ? i : (i ^ 0x7FFFFFFF); }
__device__ __forceinline__ float ordf(int i)   { return __int_as_float(i >= 0 ? i : (i ^ 0x7FFFFFFF)); }

// ---------- init / convert ----------
__global__ void k_f32_to_bf16(const float* __restrict__ in, unsigned short* __restrict__ out, int n) {
  int t = blockIdx.x * blockDim.x + threadIdx.x;
  if (t < n) out[t] = f2bf(in[t]);
}
__global__ void k_zero_f32(float* p, int n) {
  int t = blockIdx.x * blockDim.x + threadIdx.x;
  if (t < n) p[t] = 0.0f;
}
__global__ void k_set_i32(int* p, int n, int v) {
  int t = blockIdx.x * blockDim.x + threadIdx.x;
  if (t < n) p[t] = v;
}

// ---------- pack W [K][NT*16] f32 into WMMA B-fragment order (bf16) ----------
// Bfrag[((tn*KS + ks)*32 + lane)*16 + e] = bf16(W[ks*32 + (lane>>4)*8 + (e<8 ? e : e+8)][tn*16 + (lane&15)])
template<int K, int NT>
__global__ void k_pack_bfrag(const float* __restrict__ W, unsigned short* __restrict__ Bfrag) {
  constexpr int KS = K / 32;
  int t = blockIdx.x * blockDim.x + threadIdx.x;
  if (t >= NT * KS * 32) return;
  int lane = t & 31;
  int ks = (t >> 5) % KS;
  int tn = (t >> 5) / KS;
  int n = tn * 16 + (lane & 15);
  int kbase = ks * 32 + (lane >> 4) * 8;
  unsigned short* o = Bfrag + (size_t)t * 16;
#pragma unroll
  for (int e = 0; e < 16; ++e)
    o[e] = f2bf(W[(size_t)(kbase + (e < 8 ? e : e + 8)) * (NT * 16) + n]);
}

// ---------- bf16 WMMA GEMM: C[M, NT*16] = A[M, K] * B, fp32 accum ----------
// Block = NT waves; block b owns M-tile b (staged in LDS), wave w owns N-tile w.
// blockDim.x == NT*32; grid = M/16. EXEC all-1s at every wmma.
template<int K, int NT>
__global__ void k_gemm_bf16(const unsigned short* __restrict__ A,
                            const unsigned short* __restrict__ Bfrag,
                            float* __restrict__ C) {
  constexpr int KS   = K / 32;
  constexpr int PITCH = K + 8;                 // bf16 row pitch (bank spread)
  __shared__ unsigned short sA[16 * PITCH];
  const int tid  = threadIdx.x;
  const int lane = tid & 31;
  const int tn   = tid >> 5;                   // wave id = N-tile
  const int tm   = blockIdx.x;
  const int Nn   = NT * 16;

  // ---- cooperative A-tile stage: 16 rows x K bf16, 16B chunks ----
  constexpr int CPR = K / 8;                   // 16B chunks per row
  for (int c = tid; c < 16 * CPR; c += NT * 32) {
    int row = c / CPR, cc = c - row * CPR;
    uint4 v = *(const uint4*)(A + ((size_t)tm * 16 + row) * K + cc * 8);
    *(uint4*)(sA + row * PITCH + cc * 8) = v;
  }
  __syncthreads();

  const int half = lane >> 4, l15 = lane & 15;
  v8f acc = {};
  const unsigned short* arow = sA + l15 * PITCH + half * 8;            // LDS
  const unsigned short* bf   = Bfrag + ((size_t)(tn * KS) * 32 + lane) * 16;

#pragma unroll
  for (int ks = 0; ks < KS; ++ks) {
    // A fragment: two 16B LDS loads (K chunks [half*8, +8) and +16)
    uint4 lo = *(const uint4*)(arow + ks * 32);
    uint4 hi = *(const uint4*)(arow + ks * 32 + 16);
    v8u au;
    au[0] = lo.x; au[1] = lo.y; au[2] = lo.z; au[3] = lo.w;
    au[4] = hi.x; au[5] = hi.y; au[6] = hi.z; au[7] = hi.w;
    v16bf a = __builtin_bit_cast(v16bf, au);
    // B fragment: one contiguous 32B load from packed buffer
    v8u bu = *(const v8u*)(bf + (size_t)ks * (32 * 16));
    v16bf b = __builtin_bit_cast(v16bf, bu);
    acc = __builtin_amdgcn_wmma_f32_16x16x32_bf16(false, a, false, b,
                                                  (short)0, acc, false, false);
  }

  // D: lanes 0-15 -> col=lane, rows r; lanes 16-31 -> col=lane-16, rows r+8
  float* crow = C + ((size_t)tm * 16 + half * 8) * Nn + tn * 16 + l15;
#pragma unroll
  for (int r = 0; r < 8; ++r) crow[(size_t)r * Nn] = acc[r];
}

// ---------- per-node attention logits (layer 1): t = n*HEADS + h ----------
__global__ void k_alpha1(const float* __restrict__ h,
                         const float* __restrict__ asv, const float* __restrict__ adv,
                         float* __restrict__ as_o, float* __restrict__ ad_o) {
  int t = blockIdx.x * blockDim.x + threadIdx.x;
  if (t >= N_NODES * HEADS) return;
  int n = t >> 3, hh = t & 7;
  const float* hp = h + (size_t)n * H1DIM + hh * NHID;
  float s = 0.f, d = 0.f;
#pragma unroll
  for (int c = 0; c < NHID; ++c) {
    float v = hp[c];
    s += v * asv[hh * NHID + c];
    d += v * adv[hh * NHID + c];
  }
  as_o[t] = s; ad_o[t] = d;
}

// ---------- per-node attention logits (layer 2, 1 head) ----------
__global__ void k_alpha2(const float* __restrict__ h,
                         const float* __restrict__ asv, const float* __restrict__ adv,
                         float* __restrict__ as_o, float* __restrict__ ad_o) {
  int n = blockIdx.x * blockDim.x + threadIdx.x;
  if (n >= N_NODES) return;
  float s = 0.f, d = 0.f;
#pragma unroll
  for (int c = 0; c < NCLASS; ++c) {
    float v = h[(size_t)n * NCLASS + c];
    s += v * asv[c];
    d += v * adv[c];
  }
  as_o[n] = s; ad_o[n] = d;
}

__device__ __forceinline__ void edge_sd(const int* __restrict__ ei, int e, int& s, int& d) {
  if (e < N_EDGES) { s = ei[e]; d = ei[N_EDGES + e]; }
  else             { s = e - N_EDGES; d = s; }
}

// ---------- segment max of leaky-relu logits ----------
template<int H>
__global__ void k_edge_max(const int* __restrict__ ei, const float* __restrict__ as,
                           const float* __restrict__ ad, int* __restrict__ m) {
  int e = blockIdx.x * blockDim.x + threadIdx.x;
  if (e >= ETOT) return;
  int s, d; edge_sd(ei, e, s, d);
#pragma unroll
  for (int hh = 0; hh < H; ++hh) {
    float v = as[(size_t)s * H + hh] + ad[(size_t)d * H + hh];
    v = v > 0.f ? v : NEG_SLOPE * v;
    atomicMax(m + (size_t)d * H + hh, ford(v));
  }
}

// ---------- segment sum of exp(logit - max) ----------
template<int H>
__global__ void k_edge_sum(const int* __restrict__ ei, const float* __restrict__ as,
                           const float* __restrict__ ad, const int* __restrict__ m,
                           float* __restrict__ den) {
  int e = blockIdx.x * blockDim.x + threadIdx.x;
  if (e >= ETOT) return;
  int s, d; edge_sd(ei, e, s, d);
#pragma unroll
  for (int hh = 0; hh < H; ++hh) {
    float v = as[(size_t)s * H + hh] + ad[(size_t)d * H + hh];
    v = v > 0.f ? v : NEG_SLOPE * v;
    atomicAdd(den + (size_t)d * H + hh, __expf(v - ordf(m[(size_t)d * H + hh])));
  }
}

// ---------- fold q = m + log(den): softmax weight = exp(logit - q) ----------
__global__ void k_foldq(const int* __restrict__ m, float* __restrict__ den, int n) {
  int t = blockIdx.x * blockDim.x + threadIdx.x;
  if (t < n) den[t] = ordf(m[t]) + __logf(den[t]);
}

// ---------- weighted aggregation: thread per (edge, head); C channels inner ----------
template<int H, int C>
__global__ void k_aggregate(const int* __restrict__ ei, const float* __restrict__ as,
                            const float* __restrict__ ad, const float* __restrict__ q,
                            const float* __restrict__ hsrc, float* __restrict__ agg) {
  int t = blockIdx.x * blockDim.x + threadIdx.x;
  if (t >= ETOT * H) return;
  int e = t / H, hh = t - e * H;
  int s, d; edge_sd(ei, e, s, d);
  float v = as[(size_t)s * H + hh] + ad[(size_t)d * H + hh];
  v = v > 0.f ? v : NEG_SLOPE * v;
  float a = __expf(v - q[(size_t)d * H + hh]);
  const float4* hs4 = (const float4*)(hsrc + ((size_t)s * H + hh) * C);
  float*        ag  = agg + ((size_t)d * H + hh) * C;
#pragma unroll
  for (int cc = 0; cc < C / 4; ++cc) {
    float4 hv = hs4[cc];
    atomicAdd(ag + 4 * cc + 0, a * hv.x);
    atomicAdd(ag + 4 * cc + 1, a * hv.y);
    atomicAdd(ag + 4 * cc + 2, a * hv.z);
    atomicAdd(ag + 4 * cc + 3, a * hv.w);
  }
}

// ---------- layer-1 finalize: ELU(agg + b1), stored as bf16 for GEMM2 ----------
__global__ void k_fin1(const float* __restrict__ agg, const float* __restrict__ b,
                       unsigned short* __restrict__ h1b, int total) {
  int t = blockIdx.x * blockDim.x + threadIdx.x;
  if (t >= total) return;
  float v = agg[t] + b[t & (H1DIM - 1)];
  v = v > 0.f ? v : (__expf(v) - 1.0f);
  h1b[t] = f2bf(v);
}

// ---------- layer-2 finalize: out = agg + b2 ----------
__global__ void k_fin2(const float* __restrict__ agg, const float* __restrict__ b,
                       float* __restrict__ out, int total) {
  int t = blockIdx.x * blockDim.x + threadIdx.x;
  if (t >= total) return;
  out[t] = agg[t] + b[t & (NCLASS - 1)];
}

// ================= host side =================
static inline int cdiv(int a, int b) { return (a + b - 1) / b; }

extern "C" void kernel_launch(void* const* d_in, const int* in_sizes, int n_in,
                              void* d_out, int out_size, void* d_ws, size_t ws_size,
                              hipStream_t stream) {
  const float* x    = (const float*)d_in[0];
  const int*   ei   = (const int*)  d_in[1];
  const float* W1   = (const float*)d_in[2];
  const float* as1v = (const float*)d_in[3];
  const float* ad1v = (const float*)d_in[4];
  const float* b1   = (const float*)d_in[5];
  const float* W2   = (const float*)d_in[6];
  const float* as2v = (const float*)d_in[7];
  const float* ad2v = (const float*)d_in[8];
  const float* b2   = (const float*)d_in[9];
  float* out = (float*)d_out;

  // ---- carve workspace (256B aligned buffers) ----
  char* w = (char*)d_ws;
  auto alloc = [&](size_t bytes) -> char* {
    char* p = w; w += (bytes + 255) & ~(size_t)255; return p;
  };
  unsigned short* xb     = (unsigned short*)alloc((size_t)N_NODES * NFEAT * 2);
  unsigned short* bfrag1 = (unsigned short*)alloc((size_t)8 * 8 * 32 * 16 * 2);  // 64KB
  unsigned short* bfrag2 = (unsigned short*)alloc((size_t)1 * 4 * 32 * 16 * 2);  // 4KB
  float* hpre1 = (float*)alloc((size_t)N_NODES * H1DIM * 4);
  float* as1   = (float*)alloc((size_t)N_NODES * HEADS * 4);
  float* ad1   = (float*)alloc((size_t)N_NODES * HEADS * 4);
  int*   m1    = (int*)  alloc((size_t)N_NODES * HEADS * 4);
  float* den1  = (float*)alloc((size_t)N_NODES * HEADS * 4);   // becomes q1
  float* agg1  = (float*)alloc((size_t)N_NODES * H1DIM * 4);
  unsigned short* h1b = (unsigned short*)alloc((size_t)N_NODES * H1DIM * 2);
  float* h2    = (float*)alloc((size_t)N_NODES * NCLASS * 4);
  float* as2   = (float*)alloc((size_t)N_NODES * 4);
  float* ad2   = (float*)alloc((size_t)N_NODES * 4);
  int*   m2    = (int*)  alloc((size_t)N_NODES * 4);
  float* den2  = (float*)alloc((size_t)N_NODES * 4);           // becomes q2
  float* agg2  = (float*)alloc((size_t)N_NODES * NCLASS * 4);

  const int B = 256;
  const int NEG_INF_ORD = (int)0x80000000;   // INT_MIN, below ford(-inf)

  // ---- converts / fragment packing ----
  k_f32_to_bf16<<<cdiv(N_NODES * NFEAT, B), B, 0, stream>>>(x, xb, N_NODES * NFEAT);
  k_pack_bfrag<NFEAT, 8><<<cdiv(8 * 8 * 32, B), B, 0, stream>>>(W1, bfrag1);
  k_pack_bfrag<H1DIM, 1><<<cdiv(1 * 4 * 32, B), B, 0, stream>>>(W2, bfrag2);

  // ---- layer 1 ----
  k_gemm_bf16<NFEAT, 8><<<N_NODES / 16, 8 * 32, 0, stream>>>(xb, bfrag1, hpre1);
  k_alpha1<<<cdiv(N_NODES * HEADS, B), B, 0, stream>>>(hpre1, as1v, ad1v, as1, ad1);

  k_set_i32 <<<cdiv(N_NODES * HEADS, B), B, 0, stream>>>(m1,   N_NODES * HEADS, NEG_INF_ORD);
  k_zero_f32<<<cdiv(N_NODES * HEADS, B), B, 0, stream>>>(den1, N_NODES * HEADS);
  k_zero_f32<<<cdiv(N_NODES * H1DIM, B), B, 0, stream>>>(agg1, N_NODES * H1DIM);

  k_edge_max<HEADS><<<cdiv(ETOT, B), B, 0, stream>>>(ei, as1, ad1, m1);
  k_edge_sum<HEADS><<<cdiv(ETOT, B), B, 0, stream>>>(ei, as1, ad1, m1, den1);
  k_foldq<<<cdiv(N_NODES * HEADS, B), B, 0, stream>>>(m1, den1, N_NODES * HEADS);
  k_aggregate<HEADS, NHID><<<cdiv(ETOT * HEADS, B), B, 0, stream>>>(ei, as1, ad1, den1,
                                                                    hpre1, agg1);
  k_fin1<<<cdiv(N_NODES * H1DIM, B), B, 0, stream>>>(agg1, b1, h1b, N_NODES * H1DIM);

  // ---- layer 2 ----
  k_gemm_bf16<H1DIM, 1><<<N_NODES / 16, 1 * 32, 0, stream>>>(h1b, bfrag2, h2);
  k_alpha2<<<cdiv(N_NODES, B), B, 0, stream>>>(h2, as2v, ad2v, as2, ad2);

  k_set_i32 <<<cdiv(N_NODES, B), B, 0, stream>>>(m2,   N_NODES, NEG_INF_ORD);
  k_zero_f32<<<cdiv(N_NODES, B), B, 0, stream>>>(den2, N_NODES);
  k_zero_f32<<<cdiv(N_NODES * NCLASS, B), B, 0, stream>>>(agg2, N_NODES * NCLASS);

  k_edge_max<1><<<cdiv(ETOT, B), B, 0, stream>>>(ei, as2, ad2, m2);
  k_edge_sum<1><<<cdiv(ETOT, B), B, 0, stream>>>(ei, as2, ad2, m2, den2);
  k_foldq<<<cdiv(N_NODES, B), B, 0, stream>>>(m2, den2, N_NODES);
  k_aggregate<1, NCLASS><<<cdiv(ETOT, B), B, 0, stream>>>(ei, as2, ad2, den2, h2, agg2);
  k_fin2<<<cdiv(N_NODES * NCLASS, B), B, 0, stream>>>(agg2, b2, out, N_NODES * NCLASS);
}
